// MultiHeadAttention_9895604650404
// MI455X (gfx1250) — matrix operand order repaired and checked by reading, not compiled, not run on hardware
//
#include <hip/hip_runtime.h>
#include <hip/hip_bf16.h>

// MI455X / gfx1250, wave32. All matrix math via v_wmma_f32_16x16x32_bf16
// (bf16 A/B, fp32 accumulate). Compute-bound workload (~52 GFLOP on ~50MB
// of L2-resident data), so WMMA throughput is the ceiling, not HBM.

typedef __attribute__((ext_vector_type(16))) __bf16 v16bf;
typedef __attribute__((ext_vector_type(8)))  float  v8f;

#define FDIM 1024
#define HNUM 16
#define DDIM 64
#define SEQ  2048
#define BSZ  2
#define MTOT (BSZ * SEQ)   // 4096 rows
#define KCH  (FDIM / 32)   // 32 K-chunks of 32

union AFrag { v16bf v; float4 f[2]; };

__device__ inline float rmax16(float v) {
#pragma unroll
  for (int m = 1; m < 16; m <<= 1) v = fmaxf(v, __shfl_xor(v, m, 16));
  return v;
}
__device__ inline float rsum16(float v) {
#pragma unroll
  for (int m = 1; m < 16; m <<= 1) v += __shfl_xor(v, m, 16);
  return v;
}

// ---------------- fp32 -> bf16 convert ----------------
__global__ void cvt_f32_bf16(const float* __restrict__ src, __bf16* __restrict__ dst, int n) {
  int i = blockIdx.x * blockDim.x + threadIdx.x;
  if (i < n) dst[i] = (__bf16)src[i];
}

// ---------------- pack W[K][N] fp32 into WMMA B-fragment order (bf16) ----
// Fragment for (n_tile nt, k_chunk kc): lane l, elem e -> B[k][n] with
// k = kc*32 + (l/16)*16 + e, n = nt*16 + (l%16). Contiguous 32B per lane.
__global__ void pack_w(const float* __restrict__ W, __bf16* __restrict__ Wp) {
  int idx  = blockIdx.x * blockDim.x + threadIdx.x;   // 0 .. F*F-1
  int e    = idx & 15;
  int lane = (idx >> 4) & 31;
  int rest = idx >> 9;
  int kc   = rest & (KCH - 1);
  int nt   = rest / KCH;
  int k = kc * 32 + (lane >> 4) * 16 + e;
  int n = nt * 16 + (lane & 15);
  Wp[idx] = (__bf16)W[k * FDIM + n];
}

// ---------------- WMMA GEMM: [4096,1024]bf16 @ [1024,1024]packed ---------
// MODE 0: +bias, fused RoPE, write bf16 [B,H,S,D]   (Q and K)
// MODE 2: +bias, write bf16 transposed [B,H,D,S]     (V)
// MODE 3: +bias, write fp32 [M,F]                    (final output)
template <int MODE>
__global__ __launch_bounds__(256) void gemm_wmma(
    const __bf16* __restrict__ X, const __bf16* __restrict__ Wp,
    const float* __restrict__ bias, const int* __restrict__ positions,
    void* __restrict__ out) {
  const int lane  = threadIdx.x & 31;
  const int wave  = threadIdx.x >> 5;
  const int group = lane >> 4;
  const int ln    = lane & 15;
  const int nBase = blockIdx.x * 64;               // one head's 64 cols
  const int mBase = blockIdx.y * 128 + wave * 16;  // 16 rows per wave

  const __bf16* xrow = X + (size_t)(mBase + ln) * FDIM;  // A row m = ln

  v8f acc[4];
#pragma unroll
  for (int i = 0; i < 4; ++i) acc[i] = (v8f){0.f,0.f,0.f,0.f,0.f,0.f,0.f,0.f};

  for (int kc = 0; kc < KCH; ++kc) {
    AFrag a;                                     // 16x32 bf16 A fragment
    const char* ab = (const char*)xrow + kc * 64;
    a.f[0] = *(const float4*)(ab + group * 32);        // K = g*16 .. g*16+7
    a.f[1] = *(const float4*)(ab + group * 32 + 16);   // K = g*16+8 ..
#pragma unroll
    for (int dt = 0; dt < 4; ++dt) {
      const v16bf* bp = (const v16bf*)(Wp +
          (size_t)(((nBase >> 4) + dt) * KCH + kc) * 512 + lane * 16);
      acc[dt] = __builtin_amdgcn_wmma_f32_16x16x32_bf16(
          false, a.v, false, *bp, (short)0, acc[dt], false, false);
    }
  }

#pragma unroll
  for (int r = 0; r < 8; ++r) {
    const int m = mBase + r + group * 8;   // global row (C layout)
    if (MODE == 3) {
      float* o = (float*)out;
#pragma unroll
      for (int dt = 0; dt < 4; ++dt)
        o[(size_t)m * FDIM + nBase + dt * 16 + ln] =
            acc[dt][r] + bias[nBase + dt * 16 + ln];
    } else if (MODE == 2) {
      __bf16* vt = (__bf16*)out;
      const int b = m >> 11, s = m & (SEQ - 1);
      const int h = nBase >> 6;
#pragma unroll
      for (int dt = 0; dt < 4; ++dt) {
        const int d = dt * 16 + ln;
        vt[((size_t)(b * HNUM + h) * DDIM + d) * SEQ + s] =
            (__bf16)(acc[dt][r] + bias[nBase + d]);
      }
    } else {  // MODE 0: RoPE
      __bf16* q = (__bf16*)out;
      const int b = m >> 11, s = m & (SEQ - 1);
      const int h = nBase >> 6;
      const float pos = (float)positions[m];
      __bf16* row = q + ((size_t)(b * HNUM + h) * SEQ + s) * DDIM;
#pragma unroll
      for (int dt = 0; dt < 2; ++dt) {
        const int i = dt * 16 + ln;  // 0..31, partner at i+32 in acc[dt+2]
        const float x1 = acc[dt][r]     + bias[nBase + i];
        const float x2 = acc[dt + 2][r] + bias[nBase + i + 32];
        const float ang = pos * __powf(10000.f, -(float)i * (1.f / 32.f));
        float sn, cs;
        __sincosf(ang, &sn, &cs);
        row[i]      = (__bf16)(x1 * cs - x2 * sn);
        row[i + 32] = (__bf16)(x2 * cs + x1 * sn);
      }
    }
  }
}

// ---------------- flash attention: 1 wave = 16 query rows ----------------
__global__ __launch_bounds__(256) void flash_attn(
    const __bf16* __restrict__ qb, const __bf16* __restrict__ kb,
    const __bf16* __restrict__ vt, __bf16* __restrict__ att) {
  const int lane  = threadIdx.x & 31;
  const int wave  = threadIdx.x >> 5;
  const int group = lane >> 4;
  const int ln    = lane & 15;
  const int bh = blockIdx.x;                 // b*H + h
  const int b = bh >> 4, h = bh & 15;
  const int qt    = blockIdx.y * 8 + wave;
  const int qbase = qt * 16;

  __shared__ __bf16 Ps[8][16 * 32];          // per-wave prob staging (1KB)
  __bf16* myP = Ps[wave];

  // Q A-fragments, kept in registers for the whole KV loop
  const __bf16* Q = qb + ((size_t)bh * SEQ + qbase) * DDIM;
  v16bf qa[2];
  {
    const char* qrow = (const char*)(Q + (size_t)ln * DDIM);
#pragma unroll
    for (int c = 0; c < 2; ++c) {
      AFrag a;
      a.f[0] = *(const float4*)(qrow + c * 64 + group * 32);
      a.f[1] = *(const float4*)(qrow + c * 64 + group * 32 + 16);
      qa[c] = a.v;
    }
  }

  v8f o[4];
#pragma unroll
  for (int i = 0; i < 4; ++i) o[i] = (v8f){0.f,0.f,0.f,0.f,0.f,0.f,0.f,0.f};
  float mrun[8], lrun[8];
#pragma unroll
  for (int r = 0; r < 8; ++r) { mrun[r] = -3.0e38f; lrun[r] = 0.f; }

  const __bf16* Kb = kb + (size_t)bh * SEQ * DDIM;
  const __bf16* Vb = vt + (size_t)bh * DDIM * SEQ;

  const int nblocks = (qbase + 47) >> 5;  // ceil((qbase+16)/32), causal
  for (int jb = 0; jb < nblocks; ++jb) {
    const int k0 = jb * 32;
    v8f sc[2];
    sc[0] = (v8f){0.f,0.f,0.f,0.f,0.f,0.f,0.f,0.f};
    sc[1] = (v8f){0.f,0.f,0.f,0.f,0.f,0.f,0.f,0.f};
#pragma unroll
    for (int t = 0; t < 2; ++t) {          // 16-key column tiles
      const __bf16* krow = Kb + (size_t)(k0 + t * 16 + ln) * DDIM;
#pragma unroll
      for (int c = 0; c < 2; ++c) {        // d chunks of 32
        const v16bf kf = *(const v16bf*)(krow + c * 32 + group * 16);
        sc[t] = __builtin_amdgcn_wmma_f32_16x16x32_bf16(
            false, qa[c], false, kf, (short)0, sc[t], false, false);
      }
    }
    // scale, causal mask, online softmax (rows live in half-wave groups)
#pragma unroll
    for (int r = 0; r < 8; ++r) {
      const int m = qbase + r + group * 8;
      float s0 = sc[0][r] * 0.125f;        // 1/sqrt(64)
      float s1 = sc[1][r] * 0.125f;
      if (k0 + ln > m)      s0 = -3.0e38f;
      if (k0 + 16 + ln > m) s1 = -3.0e38f;
      const float tmax = rmax16(fmaxf(s0, s1));
      const float mnew  = fmaxf(mrun[r], tmax);
      const float alpha = __expf(mrun[r] - mnew);
      mrun[r] = mnew;
      const float p0 = __expf(s0 - mnew);
      const float p1 = __expf(s1 - mnew);
      lrun[r] = lrun[r] * alpha + rsum16(p0 + p1);
#pragma unroll
      for (int dt = 0; dt < 4; ++dt) o[dt][r] *= alpha;
      const int ml = r + group * 8;
      myP[ml * 32 + ln]      = (__bf16)p0;
      myP[ml * 32 + 16 + ln] = (__bf16)p1;
    }
    asm volatile("s_wait_dscnt 0" ::: "memory");
    AFrag pa;                              // P 16x32 -> A fragment
    const char* prow = (const char*)(myP + ln * 32);
    pa.f[0] = *(const float4*)(prow + group * 32);
    pa.f[1] = *(const float4*)(prow + group * 32 + 16);
#pragma unroll
    for (int dt = 0; dt < 4; ++dt) {       // O += P @ V (V transposed)
      const v16bf vf = *(const v16bf*)(Vb + (size_t)(dt * 16 + ln) * SEQ +
                                       k0 + group * 16);
      o[dt] = __builtin_amdgcn_wmma_f32_16x16x32_bf16(
          false, pa.v, false, vf, (short)0, o[dt], false, false);
    }
  }

  // normalize and write bf16 [B*S, F] for the output projection
#pragma unroll
  for (int r = 0; r < 8; ++r) {
    const int s = qbase + r + group * 8;
    const float inv = 1.f / lrun[r];
#pragma unroll
    for (int dt = 0; dt < 4; ++dt)
      att[((size_t)(b * SEQ + s)) * FDIM + h * DDIM + dt * 16 + ln] =
          (__bf16)(o[dt][r] * inv);
  }
}

extern "C" void kernel_launch(void* const* d_in, const int* in_sizes, int n_in,
                              void* d_out, int out_size, void* d_ws, size_t ws_size,
                              hipStream_t stream) {
  (void)in_sizes; (void)n_in; (void)out_size; (void)ws_size;
  const float* x  = (const float*)d_in[0];
  const int* positions = (const int*)d_in[1];
  const float* Wq = (const float*)d_in[2];
  const float* bq = (const float*)d_in[3];
  const float* Wk = (const float*)d_in[4];
  const float* bk = (const float*)d_in[5];
  const float* Wv = (const float*)d_in[6];
  const float* bv = (const float*)d_in[7];
  const float* Wo = (const float*)d_in[8];
  const float* bo = (const float*)d_in[9];
  float* out = (float*)d_out;

  char* ws = (char*)d_ws;
  const size_t MB = (size_t)1 << 20;
  __bf16* xb  = (__bf16*)(ws);             // 8 MB  x bf16 [4096,1024]
  __bf16* wpq = (__bf16*)(ws +  8 * MB);   // 2 MB  packed weights
  __bf16* wpk = (__bf16*)(ws + 10 * MB);
  __bf16* wpv = (__bf16*)(ws + 12 * MB);
  __bf16* wpo = (__bf16*)(ws + 14 * MB);
  __bf16* qb  = (__bf16*)(ws + 16 * MB);   // 8 MB  [B,H,S,D]
  __bf16* kbf = (__bf16*)(ws + 24 * MB);   // 8 MB  [B,H,S,D]
  __bf16* vt  = (__bf16*)(ws + 32 * MB);   // 8 MB  [B,H,D,S]
  __bf16* att = (__bf16*)(ws + 40 * MB);   // 8 MB  [B*S,F]

  const int NX = MTOT * FDIM;  // 4Mi elements
  cvt_f32_bf16<<<dim3(NX / 256), dim3(256), 0, stream>>>(x, xb, NX);
  pack_w<<<dim3((FDIM * FDIM) / 256), dim3(256), 0, stream>>>(Wq, wpq);
  pack_w<<<dim3((FDIM * FDIM) / 256), dim3(256), 0, stream>>>(Wk, wpk);
  pack_w<<<dim3((FDIM * FDIM) / 256), dim3(256), 0, stream>>>(Wv, wpv);
  pack_w<<<dim3((FDIM * FDIM) / 256), dim3(256), 0, stream>>>(Wo, wpo);

  dim3 ggrid(FDIM / 64, MTOT / 128);       // (16, 32), 256 thr = 8 waves
  gemm_wmma<0><<<ggrid, 256, 0, stream>>>(xb, wpq, bq, positions, qb);
  gemm_wmma<0><<<ggrid, 256, 0, stream>>>(xb, wpk, bk, positions, kbf);
  gemm_wmma<2><<<ggrid, 256, 0, stream>>>(xb, wpv, bv, positions, vt);

  flash_attn<<<dim3(BSZ * HNUM, SEQ / 128), 256, 0, stream>>>(qb, kbf, vt, att);

  gemm_wmma<3><<<ggrid, 256, 0, stream>>>(att, wpo, bo, positions, out);
}
